// SelfAttentionBlock_51419348468291
// MI455X (gfx1250) — matrix operand order repaired and checked by reading, not compiled, not run on hardware
//
#include <hip/hip_runtime.h>
#include <math.h>

#define D_    384
#define H_    6
#define DK_   64
#define NSEQ  2048
#define B_    8
#define BN    (B_ * NSEQ)      // 16384 rows
#define CH    (3 + D_)         // 387

typedef __attribute__((ext_vector_type(16))) _Float16 v16h;
typedef __attribute__((ext_vector_type(8)))  _Float16 v8h;
typedef __attribute__((ext_vector_type(8)))  float    v8f;

#define WMMA_F16(a, b, c) \
  __builtin_amdgcn_wmma_f32_16x16x32_f16(false, (a), false, (b), (short)0, (c), false, false)

#if defined(__has_builtin)
#if __has_builtin(__builtin_amdgcn_tensor_load_to_lds) && __has_builtin(__builtin_amdgcn_s_wait_tensorcnt)
#define USE_TDM 1
#endif
#endif
#ifndef USE_TDM
#define USE_TDM 0
#endif

#if USE_TDM
typedef unsigned int tdm_v4u __attribute__((ext_vector_type(4)));
typedef int          tdm_v8i __attribute__((ext_vector_type(8)));
typedef int          tdm_v4i __attribute__((ext_vector_type(4)));

// Tensor DMA descriptor per CDNA5 ISA ch.8: group0 (count/lds_addr/global_addr/type),
// group1 (data_size, tensor dims, tile dims, dim0 stride). 2-byte elements.
// 1D copy: tile_d1 == 0. 2D tile: tile_d1 rows of tile_d0 elems, row stride = stride0.
__device__ inline void tdm_load_f16(const void* gptr, unsigned lds_byte_addr,
                                    unsigned tile_d0, unsigned tile_d1,
                                    unsigned long long stride0,
                                    unsigned tensor_d0, unsigned tensor_d1) {
  const unsigned long long ga = (unsigned long long)(uintptr_t)gptr;
  tdm_v4u g0;
  g0[0] = 1u;                                              // count=1, user descriptor
  g0[1] = lds_byte_addr;                                   // lds_addr [63:32]
  g0[2] = (unsigned)(ga & 0xFFFFFFFFu);                    // global_addr [95:64]
  g0[3] = (unsigned)((ga >> 32) & 0x1FFFFFFu) | (2u << 30);// global_addr[56:32] | type=2
  tdm_v8i g1;
  g1[0] = (int)(1u << 16);                                 // data_size=1 (2 bytes)
  g1[1] = (int)((tensor_d0 & 0xFFFFu) << 16);              // tensor_dim0[15:0] -> bits 63:48
  g1[2] = (int)((tensor_d0 >> 16) | ((tensor_d1 & 0xFFFFu) << 16));
  g1[3] = (int)((tensor_d1 >> 16) | ((tile_d0 & 0xFFFFu) << 16));
  g1[4] = (int)(tile_d1 & 0xFFFFu);                        // tile_dim1; tile_dim2=0
  g1[5] = (int)(stride0 & 0xFFFFFFFFu);                    // tensor_dim0_stride[31:0]
  g1[6] = (int)((stride0 >> 32) & 0xFFFFu);                // stride[47:32]; dim1_stride=0
  g1[7] = 0;
  tdm_v4i g2 = {0, 0, 0, 0};
  tdm_v4i g3 = {0, 0, 0, 0};
  tdm_v8i g4 = {0, 0, 0, 0, 0, 0, 0, 0};                   // 6-arg toolchain form
  __builtin_amdgcn_tensor_load_to_lds(g0, g1, g2, g3, g4, 0);
}
__device__ inline unsigned lds_addr_of(const void* p) { return (unsigned)(uintptr_t)p; }
#endif

// ---------------- DPP 16-lane-half butterfly reductions -----------------------------
// xor1 = quad_perm(1,0,3,2); xor2 = quad_perm(2,3,0,1); then (quads uniform)
// row_half_mirror acts as xor4 and row_mirror as xor8. Never crosses a 16-lane half.
__device__ inline float dpp_max16(float x) {
  int t;
  t = __builtin_amdgcn_update_dpp(0, __float_as_int(x), 0xB1, 0xF, 0xF, true);
  x = fmaxf(x, __int_as_float(t));
  t = __builtin_amdgcn_update_dpp(0, __float_as_int(x), 0x4E, 0xF, 0xF, true);
  x = fmaxf(x, __int_as_float(t));
  t = __builtin_amdgcn_update_dpp(0, __float_as_int(x), 0x141, 0xF, 0xF, true);
  x = fmaxf(x, __int_as_float(t));
  t = __builtin_amdgcn_update_dpp(0, __float_as_int(x), 0x140, 0xF, 0xF, true);
  x = fmaxf(x, __int_as_float(t));
  return x;
}
__device__ inline float dpp_sum16(float x) {
  int t;
  t = __builtin_amdgcn_update_dpp(0, __float_as_int(x), 0xB1, 0xF, 0xF, true);
  x += __int_as_float(t);
  t = __builtin_amdgcn_update_dpp(0, __float_as_int(x), 0x4E, 0xF, 0xF, true);
  x += __int_as_float(t);
  t = __builtin_amdgcn_update_dpp(0, __float_as_int(x), 0x141, 0xF, 0xF, true);
  x += __int_as_float(t);
  t = __builtin_amdgcn_update_dpp(0, __float_as_int(x), 0x140, 0xF, 0xF, true);
  x += __int_as_float(t);
  return x;
}

// ---------------- fragment loaders (wave32, contiguous k = 16*hi + s in both A and B) ------------
__device__ inline v16h load_a16x32(const _Float16* __restrict__ A, int ld, int lane) {
  const int m = lane & 15, hi = lane >> 4;
  const _Float16* r = A + (size_t)m * ld + hi * 16;
  const v8h lo = *(const v8h*)(r);
  const v8h hh = *(const v8h*)(r + 8);
  return __builtin_shufflevector(lo, hh, 0, 1, 2, 3, 4, 5, 6, 7, 8, 9, 10, 11, 12, 13, 14, 15);
}
__device__ inline v16h load_bT32x16(const _Float16* __restrict__ W, int ld, int lane) {
  const int n = lane & 15, hi = lane >> 4;
  const _Float16* r = W + (size_t)n * ld + hi * 16;
  const v8h lo = *(const v8h*)(r);
  const v8h hh = *(const v8h*)(r + 8);
  return __builtin_shufflevector(lo, hh, 0, 1, 2, 3, 4, 5, 6, 7, 8, 9, 10, 11, 12, 13, 14, 15);
}

// ---------------- LayerNorm kernels -------------------------------------------------

__global__ __launch_bounds__(128) void ln1_prep_kernel(
    const float* __restrict__ pts, const float* __restrict__ g, const float* __restrict__ be,
    float* __restrict__ feats_res, _Float16* __restrict__ h16, float* __restrict__ dout) {
  const int row = blockIdx.x;                // b*NSEQ + n
  const int b = row / NSEQ, n = row % NSEQ;
  const float* base = pts + ((size_t)b * CH + 3) * NSEQ + n;
  float x[3];
  float s = 0.f, s2 = 0.f;
#pragma unroll
  for (int k = 0; k < 3; ++k) {
    const int d = threadIdx.x + 128 * k;
    const float v = base[(size_t)d * NSEQ];
    x[k] = v; s += v; s2 += v * v;
  }
#pragma unroll
  for (int m = 16; m >= 1; m >>= 1) { s += __shfl_xor(s, m, 32); s2 += __shfl_xor(s2, m, 32); }
  __shared__ float rs[4], rs2[4];
  const int w = threadIdx.x >> 5;
  if ((threadIdx.x & 31) == 0) { rs[w] = s; rs2[w] = s2; }
  __syncthreads();
  s  = rs[0] + rs[1] + rs[2] + rs[3];
  s2 = rs2[0] + rs2[1] + rs2[2] + rs2[3];
  const float mu   = s * (1.0f / D_);
  const float var  = s2 * (1.0f / D_) - mu * mu;
  const float rstd = rsqrtf(var + 1e-5f);
#pragma unroll
  for (int k = 0; k < 3; ++k) {
    const int d = threadIdx.x + 128 * k;
    const float v = x[k];
    feats_res[(size_t)row * D_ + d] = v;
    h16[(size_t)row * D_ + d] = (_Float16)((v - mu) * rstd * g[d] + be[d]);
  }
  if (threadIdx.x < 3)
    dout[((size_t)b * CH + threadIdx.x) * NSEQ + n] = pts[((size_t)b * CH + threadIdx.x) * NSEQ + n];
}

__global__ __launch_bounds__(128) void ln2_kernel(
    const float* __restrict__ xin, const float* __restrict__ g, const float* __restrict__ be,
    _Float16* __restrict__ h16) {
  const int row = blockIdx.x;
  const float* base = xin + (size_t)row * D_;
  float x[3];
  float s = 0.f, s2 = 0.f;
#pragma unroll
  for (int k = 0; k < 3; ++k) {
    const int d = threadIdx.x + 128 * k;
    const float v = base[d];
    x[k] = v; s += v; s2 += v * v;
  }
#pragma unroll
  for (int m = 16; m >= 1; m >>= 1) { s += __shfl_xor(s, m, 32); s2 += __shfl_xor(s2, m, 32); }
  __shared__ float rs[4], rs2[4];
  const int w = threadIdx.x >> 5;
  if ((threadIdx.x & 31) == 0) { rs[w] = s; rs2[w] = s2; }
  __syncthreads();
  s  = rs[0] + rs[1] + rs[2] + rs[3];
  s2 = rs2[0] + rs2[1] + rs2[2] + rs2[3];
  const float mu   = s * (1.0f / D_);
  const float var  = s2 * (1.0f / D_) - mu * mu;
  const float rstd = rsqrtf(var + 1e-5f);
#pragma unroll
  for (int k = 0; k < 3; ++k) {
    const int d = threadIdx.x + 128 * k;
    h16[(size_t)row * D_ + d] = (_Float16)((x[k] - mu) * rstd * g[d] + be[d]);
  }
}

__global__ void f32_to_f16_kernel(const float* __restrict__ src, _Float16* __restrict__ dst, int n) {
  const int i = blockIdx.x * 256 + threadIdx.x;
  if (i < n) dst[i] = (_Float16)src[i];
}

// ---------------- LDS-staged WMMA GEMM ----------------------------------------------
// One block owns one 32-row A tile (TDM-staged into LDS); 8 waves sweep column tiles.
// C = A * W^T (W row-major [Ncol x K]), f16 in / f32 accumulate.
template <int MODE>
__global__ __launch_bounds__(256) void gemm32(
    const _Float16* __restrict__ A, const _Float16* __restrict__ W,
    const float* __restrict__ bias, const float* __restrict__ resid,
    float* __restrict__ outf, _Float16* __restrict__ outh,
    _Float16* __restrict__ q16, _Float16* __restrict__ k16o, _Float16* __restrict__ vt16,
    int K, int Ncol) {
  __shared__ __align__(16) _Float16 Alds[32 * 768];   // up to 48 KB
  const int lane = threadIdx.x & 31;
  const int waveId = threadIdx.x >> 5;
  const int row0 = blockIdx.x * 32;
  const int total = 32 * K;

#if USE_TDM
  if (waveId == 0) {
    tdm_load_f16(A + (size_t)row0 * K, lds_addr_of(&Alds[0]),
                 (unsigned)total, 0u, (unsigned long long)total, (unsigned)total, 0u);
    __builtin_amdgcn_s_wait_tensorcnt(0);
  }
#else
  for (int ofs = threadIdx.x * 8; ofs < total; ofs += 256 * 8)
    *(v8h*)(Alds + ofs) = *(const v8h*)(A + (size_t)row0 * K + ofs);
#endif
  __syncthreads();

  const int tilesN = Ncol >> 5;
  const int hi = lane >> 4, nn = lane & 15;
  for (int tn = waveId; tn < tilesN; tn += 8) {
    const int col0 = tn * 32;
    v8f acc00 = {}, acc01 = {}, acc10 = {}, acc11 = {};
    for (int k0 = 0; k0 < K; k0 += 32) {
      if (k0 + 32 < K) __builtin_prefetch(W + (size_t)col0 * K + k0 + 32, 0, 0);
      const v16h a0 = load_a16x32(Alds + k0, K, lane);
      const v16h a1 = load_a16x32(Alds + 16 * K + k0, K, lane);
      const v16h b0 = load_bT32x16(W + (size_t)col0 * K + k0, K, lane);
      const v16h b1 = load_bT32x16(W + (size_t)(col0 + 16) * K + k0, K, lane);
      acc00 = WMMA_F16(a0, b0, acc00);
      acc01 = WMMA_F16(a0, b1, acc01);
      acc10 = WMMA_F16(a1, b0, acc10);
      acc11 = WMMA_F16(a1, b1, acc11);
    }
#pragma unroll
    for (int i = 0; i < 2; ++i) {
#pragma unroll
      for (int j = 0; j < 2; ++j) {
        const v8f acc = (i == 0) ? (j == 0 ? acc00 : acc01) : (j == 0 ? acc10 : acc11);
#pragma unroll
        for (int r = 0; r < 8; ++r) {
          const int row = row0 + i * 16 + r + 8 * hi;
          const int col = col0 + j * 16 + nn;
          float val = acc[r];
          if (MODE == 0) {
            const int which = col / D_;
            const int e = col % D_;
            const int hh = e >> 6, d = e & 63;
            const int b = row >> 11, n = row & (NSEQ - 1);
            if (which == 0) val *= 0.125f;  // DK^-0.5 folded into Q
            if (which == 2) {
              vt16[((size_t)(b * H_ + hh) * DK_ + d) * NSEQ + n] = (_Float16)val;  // VT[d][n]
            } else {
              _Float16* dst = (which == 0) ? q16 : k16o;
              dst[((size_t)(b * H_ + hh) * NSEQ + n) * DK_ + d] = (_Float16)val;
            }
          } else if (MODE == 1) {
            val += bias[col] + resid[(size_t)row * D_ + col];
            outf[(size_t)row * D_ + col] = val;
          } else if (MODE == 2) {
            const float t = val + bias[col];
            outh[(size_t)row * (2 * D_) + col] = (_Float16)(0.5f * t * (1.0f + erff(t * 0.70710678118f)));
          } else {
            const float t = val + bias[col] + resid[(size_t)row * D_ + col];
            const int b = row >> 11, n = row & (NSEQ - 1);
            outf[((size_t)b * CH + 3 + col) * NSEQ + n] = t;
          }
        }
      }
    }
  }
}

// ---------------- flash attention ---------------------------------------------------
// Block = 8 waves sharing one (b,h); each wave owns one 16-row Q tile (128 rows/block).
// K and VT tiles are double-buffered in LDS, staged by the Tensor Data Mover,
// overlapping the DMA for tile j+1 with compute on tile j.
__global__ __launch_bounds__(256) void attn_kernel(
    const _Float16* __restrict__ q16, const _Float16* __restrict__ k16,
    const _Float16* __restrict__ vt16, _Float16* __restrict__ ctx16) {
  __shared__ __align__(16) _Float16 Klds[2][32 * DK_];   // 2 x 4 KB
  __shared__ __align__(16) _Float16 VTlds[2][DK_ * 32];  // 2 x 4 KB
  __shared__ __align__(16) _Float16 Plds[8][16 * 32];    // 8 KB
  const int lane = threadIdx.x & 31;
  const int waveId = threadIdx.x >> 5;
  const int bh = blockIdx.x >> 4;                 // [0, 48)
  const int qblk = blockIdx.x & 15;
  const _Float16* Q  = q16  + (size_t)bh * NSEQ * DK_;
  const _Float16* Km = k16  + (size_t)bh * NSEQ * DK_;
  const _Float16* VT = vt16 + (size_t)bh * NSEQ * DK_;   // [DK_][NSEQ]
  const int i0 = qblk * 128 + waveId * 16;
  const int hi = lane >> 4, nn = lane & 15;

  const v16h aq0 = load_a16x32(Q + (size_t)i0 * DK_ + 0, DK_, lane);
  const v16h aq1 = load_a16x32(Q + (size_t)i0 * DK_ + 32, DK_, lane);

  v8f acc0 = {}, acc1 = {}, acc2 = {}, acc3 = {};
  float mrow[8], lrow[8];
#pragma unroll
  for (int r = 0; r < 8; ++r) { mrow[r] = -1e30f; lrow[r] = 0.f; }
  _Float16* Pw = &Plds[waveId][0];

  auto stage = [&](int j0, int buf) {
#if USE_TDM
    if (waveId == 0) {
      // K tile: rows j0..j0+31 of (NSEQ x DK_) are one contiguous 32*64 block (1D)
      tdm_load_f16(Km + (size_t)j0 * DK_, lds_addr_of(&Klds[buf][0]),
                   32u * DK_, 0u, 32ull * DK_, 32u * DK_, 0u);
      // VT tile: 64 rows x 32 cols, row stride NSEQ (2D)
      tdm_load_f16(VT + (size_t)j0, lds_addr_of(&VTlds[buf][0]),
                   32u, (unsigned)DK_, (unsigned long long)NSEQ, (unsigned)NSEQ, (unsigned)DK_);
    }
#else
    {
      const int t = threadIdx.x;
      *(v8h*)(&Klds[buf][t * 8]) = *(const v8h*)(Km + (size_t)j0 * DK_ + t * 8);
      const int rowd = t >> 2, seg = (t & 3) * 8;
      *(v8h*)(&VTlds[buf][rowd * 32 + seg]) = *(const v8h*)(VT + (size_t)rowd * NSEQ + j0 + seg);
    }
#endif
  };

  stage(0, 0);
  for (int it = 0; it < NSEQ / 32; ++it) {
    const int j0 = it * 32;
    const int buf = it & 1;
#if USE_TDM
    if (waveId == 0) __builtin_amdgcn_s_wait_tensorcnt(0);
#endif
    __syncthreads();                         // staged tiles visible to all waves
    if (j0 + 32 < NSEQ) stage(j0 + 32, buf ^ 1);

    // scores S = Q * K^T (Q pre-scaled): load all four K fragments up front so the
    // ds_load_b128s issue as one burst with staggered dscnt waits, then 4 WMMAs.
    const v16h bk00 = load_bT32x16(&Klds[buf][0], DK_, lane);
    const v16h bk01 = load_bT32x16(&Klds[buf][32], DK_, lane);
    const v16h bk10 = load_bT32x16(&Klds[buf][16 * DK_ + 0], DK_, lane);
    const v16h bk11 = load_bT32x16(&Klds[buf][16 * DK_ + 32], DK_, lane);
    v8f s0 = {}, s1 = {};
    s0 = WMMA_F16(aq0, bk00, s0);
    s0 = WMMA_F16(aq1, bk01, s0);
    s1 = WMMA_F16(aq0, bk10, s1);
    s1 = WMMA_F16(aq1, bk11, s1);

    // online softmax per accumulator row (row = r + 8*hi); DPP 16-lane-half reductions
#pragma unroll
    for (int r = 0; r < 8; ++r) {
      const float v0 = s0[r], v1 = s1[r];
      const float mx = dpp_max16(fmaxf(v0, v1));
      const float mnew = fmaxf(mrow[r], mx);
      const float corr = __expf(mrow[r] - mnew);
      const float p0 = __expf(v0 - mnew);
      const float p1 = __expf(v1 - mnew);
      const float psum = dpp_sum16(p0 + p1);
      lrow[r] = lrow[r] * corr + psum;
      mrow[r] = mnew;
      acc0[r] *= corr; acc1[r] *= corr; acc2[r] *= corr; acc3[r] *= corr;
      Pw[(r + 8 * hi) * 32 + nn]      = (_Float16)p0;
      Pw[(r + 8 * hi) * 32 + nn + 16] = (_Float16)p1;
    }

    // ctx += P * V  (B[k][n] = VT[d0+n][j0+k]); grouped loads again
    const v16h ap  = load_a16x32(Pw, 32, lane);
    const v16h bv0 = load_bT32x16(&VTlds[buf][0 * 32], 32, lane);
    const v16h bv1 = load_bT32x16(&VTlds[buf][16 * 32], 32, lane);
    const v16h bv2 = load_bT32x16(&VTlds[buf][32 * 32], 32, lane);
    const v16h bv3 = load_bT32x16(&VTlds[buf][48 * 32], 32, lane);
    acc0 = WMMA_F16(ap, bv0, acc0);
    acc1 = WMMA_F16(ap, bv1, acc1);
    acc2 = WMMA_F16(ap, bv2, acc2);
    acc3 = WMMA_F16(ap, bv3, acc3);
    __syncthreads();                         // all waves done with buf before reuse
  }

  // normalize and scatter ctx back to (B, N, D) row-major with heads concatenated
  const int b = bh / H_, h = bh % H_;
#pragma unroll
  for (int r = 0; r < 8; ++r) {
    const size_t rowbase = ((size_t)(b * NSEQ + i0 + r + 8 * hi)) * D_ + h * DK_ + nn;
    const float inv = 1.0f / lrow[r];
    ctx16[rowbase + 0]  = (_Float16)(acc0[r] * inv);
    ctx16[rowbase + 16] = (_Float16)(acc1[r] * inv);
    ctx16[rowbase + 32] = (_Float16)(acc2[r] * inv);
    ctx16[rowbase + 48] = (_Float16)(acc3[r] * inv);
  }
}

// ---------------- host side ---------------------------------------------------------

extern "C" void kernel_launch(void* const* d_in, const int* in_sizes, int n_in,
                              void* d_out, int out_size, void* d_ws, size_t ws_size,
                              hipStream_t stream) {
  const float* points = (const float*)d_in[0];
  const float* ln1_g  = (const float*)d_in[1];
  const float* ln1_b  = (const float*)d_in[2];
  const float* w_qkv  = (const float*)d_in[3];
  const float* w_o    = (const float*)d_in[4];
  const float* b_o    = (const float*)d_in[5];
  const float* ln2_g  = (const float*)d_in[6];
  const float* ln2_b  = (const float*)d_in[7];
  const float* w1     = (const float*)d_in[8];
  const float* b1     = (const float*)d_in[9];
  const float* w2     = (const float*)d_in[10];
  const float* b2     = (const float*)d_in[11];
  float* out = (float*)d_out;

  char* p = (char*)d_ws;
  auto alloc = [&](size_t bytes) -> void* {
    void* r = (void*)p;
    p += (bytes + 255) & ~(size_t)255;
    return r;
  };
  _Float16* wq16 = (_Float16*)alloc((size_t)3 * D_ * D_ * 2);
  _Float16* wo16 = (_Float16*)alloc((size_t)D_ * D_ * 2);
  _Float16* w116 = (_Float16*)alloc((size_t)2 * D_ * D_ * 2);
  _Float16* w216 = (_Float16*)alloc((size_t)D_ * 2 * D_ * 2);
  float*    fres = (float*)   alloc((size_t)BN * D_ * 4);
  _Float16* h16  = (_Float16*)alloc((size_t)BN * D_ * 2);
  _Float16* q16  = (_Float16*)alloc((size_t)BN * D_ * 2);
  _Float16* k16  = (_Float16*)alloc((size_t)BN * D_ * 2);
  _Float16* vt16 = (_Float16*)alloc((size_t)BN * D_ * 2);
  _Float16* c16  = (_Float16*)alloc((size_t)BN * D_ * 2);
  float*    xbuf = (float*)   alloc((size_t)BN * D_ * 4);
  _Float16* h216 = (_Float16*)alloc((size_t)BN * D_ * 2);
  _Float16* ff16 = (_Float16*)alloc((size_t)BN * 2 * D_ * 2);

  {
    int n = 3 * D_ * D_;
    f32_to_f16_kernel<<<(n + 255) / 256, 256, 0, stream>>>(w_qkv, wq16, n);
    n = D_ * D_;
    f32_to_f16_kernel<<<(n + 255) / 256, 256, 0, stream>>>(w_o, wo16, n);
    n = 2 * D_ * D_;
    f32_to_f16_kernel<<<(n + 255) / 256, 256, 0, stream>>>(w1, w116, n);
    n = D_ * 2 * D_;
    f32_to_f16_kernel<<<(n + 255) / 256, 256, 0, stream>>>(w2, w216, n);
  }

  ln1_prep_kernel<<<BN, 128, 0, stream>>>(points, ln1_g, ln1_b, fres, h16, out);

  gemm32<0><<<BN / 32, 256, 0, stream>>>(h16, wq16, nullptr, nullptr, nullptr, nullptr,
                                         q16, k16, vt16, D_, 3 * D_);

  attn_kernel<<<48 * 16, 256, 0, stream>>>(q16, k16, vt16, c16);

  gemm32<1><<<BN / 32, 256, 0, stream>>>(c16, wo16, b_o, fres, xbuf, nullptr,
                                         nullptr, nullptr, nullptr, D_, D_);

  ln2_kernel<<<BN, 128, 0, stream>>>(xbuf, ln2_g, ln2_b, h216);

  gemm32<2><<<BN / 32, 256, 0, stream>>>(h216, w116, b1, nullptr, nullptr, ff16,
                                         nullptr, nullptr, nullptr, D_, 2 * D_);

  gemm32<3><<<BN / 32, 256, 0, stream>>>(ff16, w216, b2, xbuf, out, nullptr,
                                         nullptr, nullptr, nullptr, 2 * D_, D_);

  (void)in_sizes; (void)n_in; (void)out_size; (void)ws_size;
}
